// ReservoirLayer_55808805044442
// MI455X (gfx1250) — matrix operand order repaired and checked by reading, not compiled, not run on hardware
//
#include <hip/hip_runtime.h>
#include <stdint.h>

// ---------------------------------------------------------------------------
// Reservoir recurrence on MI455X (gfx1250):
//   S0 = pad(x, [2048, 4096]);  repeat 16x: S = leaky_relu(S @ W, 0.1)
// One-time: W -> bf16 transposed Wt[n][k] (32 MB, L2-resident), x -> bf16 S0.
// Then 16 launches of a WMMA bf16 GEMM (f32 accumulate) with fused leaky-ReLU.
// Fragments load straight from global memory in the exact CDNA5 wave32 VGPR
// layouts.  All GEMM-side pointers are pinned to address space 1 so loads
// lower to global_load_b128 (LOADcnt only) instead of flat_load_b128
// (LOADcnt+DScnt).  K loop manually unrolled by 4 with immediate offsets.
// ---------------------------------------------------------------------------

#define NDIM   4096
#define NBATCH 2048
#define INDIM  512
#define TSTEPS 16
#define NEG_SLOPE 0.1f

typedef __attribute__((ext_vector_type(16))) __bf16 v16bf;
typedef __attribute__((ext_vector_type(8)))  __bf16 v8bf;
typedef __attribute__((ext_vector_type(8)))  float  v8f;

// Global (AS1) typedefs: keeps every derived access on the global-memory path.
typedef __attribute__((address_space(1))) const __bf16 gcbf16;
typedef __attribute__((address_space(1))) __bf16       gbf16;
typedef __attribute__((address_space(1))) float        gf32;
typedef __attribute__((address_space(1))) const v8bf   gcv8bf;

union Frag16 {
    v16bf v;
    v8bf  h[2];
};

// --- Prep 1: Wt[n*N + k] = (bf16) W[k*N + n]  (tiled transpose) -------------
__global__ __launch_bounds__(256) void prep_wt(const float* __restrict__ W,
                                               __bf16* __restrict__ Wt) {
    __shared__ float tile[32][33];
    const int bx = blockIdx.x * 32;   // n base (columns of W)
    const int by = blockIdx.y * 32;   // k base (rows of W)
    const int tx = threadIdx.x;       // 0..31
    const int ty = threadIdx.y;       // 0..7
#pragma unroll
    for (int i = 0; i < 32; i += 8)
        tile[ty + i][tx] = W[(size_t)(by + ty + i) * NDIM + (bx + tx)];
    __syncthreads();
#pragma unroll
    for (int i = 0; i < 32; i += 8)
        Wt[(size_t)(bx + ty + i) * NDIM + (by + tx)] =
            (__bf16)tile[tx][ty + i];
}

// --- Prep 2: S0 = pad(x) in bf16 -------------------------------------------
__global__ __launch_bounds__(256) void prep_s0(const float* __restrict__ x,
                                               __bf16* __restrict__ S0) {
    const int idx = blockIdx.x * blockDim.x + threadIdx.x; // over 2048*4096
    const int b = idx >> 12;
    const int k = idx & (NDIM - 1);
    float v = (k < INDIM) ? x[(size_t)b * INDIM + k] : 0.0f;
    S0[idx] = (__bf16)v;
}

// --- Main step: Sout = leaky_relu(Sin @ Wt^T) -------------------------------
// Block: 256 threads = 8 waves (2 M x 4 N).  Wave tile: 64 M x 64 N
// (4x4 accumulators of v8f).  K loop in steps of 32 (one WMMA depth),
// unrolled 4x with immediate byte offsets.
__global__ __launch_bounds__(256) void reservoir_step(
    const __bf16* __restrict__ Sin,   // [2048, 4096] row-major bf16
    const __bf16* __restrict__ Wt,    // [4096(n), 4096(k)] row-major bf16
    __bf16* __restrict__ Sout,        // [2048, 4096] bf16 (next state)
    float*  __restrict__ outf)        // optional f32 output (final step)
{
    const int lane   = threadIdx.x & 31;
    const int wave   = threadIdx.x >> 5;
    const int waveM  = wave >> 2;          // 0..1
    const int waveN  = wave & 3;           // 0..3
    const int laneHi = (lane >> 4) & 1;    // 0: lanes 0-15, 1: lanes 16-31
    const int lm     = lane & 15;

    const int m0 = blockIdx.y * 128 + waveM * 64;
    const int n0 = blockIdx.x * 256 + waveN * 64;

    // Pin bases to global address space (AS1) once; all derived pointers stay
    // global so loads/stores lower to global_load/store (not flat).
    gcbf16* SinG  = (gcbf16*)(uintptr_t)Sin;
    gcbf16* WtG   = (gcbf16*)(uintptr_t)Wt;
    gbf16*  SoutG = (gbf16*)(uintptr_t)Sout;
    gf32*   outG  = (gf32*)(uintptr_t)outf;

    // A fragment pointers (per 16-row M tile):
    //   lane holds row m = m0+mt*16+lm; K chunks {kk+laneHi*8 .. +7} and +16.
    gcbf16* arow[4];
#pragma unroll
    for (int mt = 0; mt < 4; ++mt)
        arow[mt] = SinG + (size_t)(m0 + mt * 16 + lm) * NDIM + laneHi * 8;

    // B fragment pointers (per 16-col N tile):
    //   lane holds column n = n0+nt*16+lm; K chunks {kk+laneHi*8 .. +7} and +16.
    gcbf16* brow[4];
#pragma unroll
    for (int nt = 0; nt < 4; ++nt)
        brow[nt] = WtG + (size_t)(n0 + nt * 16 + lm) * NDIM + laneHi * 8;

    v8f acc[4][4] = {};

#pragma unroll 1
    for (int kk = 0; kk < NDIM; kk += 128) {   // 4 WMMA-depth steps per iter
#pragma unroll
        for (int u = 0; u < 4; ++u) {
            const int kb = u * 32;             // immediate element offset
            Frag16 a[4], b[4];
#pragma unroll
            for (int mt = 0; mt < 4; ++mt) {
                a[mt].h[0] = *(gcv8bf*)(arow[mt] + kb);
                a[mt].h[1] = *(gcv8bf*)(arow[mt] + kb + 16);
            }
#pragma unroll
            for (int nt = 0; nt < 4; ++nt) {
                b[nt].h[0] = *(gcv8bf*)(brow[nt] + kb);
                b[nt].h[1] = *(gcv8bf*)(brow[nt] + kb + 16);
            }
#pragma unroll
            for (int mt = 0; mt < 4; ++mt)
#pragma unroll
                for (int nt = 0; nt < 4; ++nt)
                    acc[mt][nt] = __builtin_amdgcn_wmma_f32_16x16x32_bf16(
                        false, a[mt].v, false, b[nt].v,
                        (short)0, acc[mt][nt], false, false);
        }
        // One pointer bump per 4 k-steps (amortized address VALU).
#pragma unroll
        for (int mt = 0; mt < 4; ++mt) arow[mt] += 128;
#pragma unroll
        for (int nt = 0; nt < 4; ++nt) brow[nt] += 128;
    }

    // Epilogue: leaky-ReLU, store bf16 next-state (+ optional f32 output).
    const int mrow = laneHi * 8;
    const bool wantF32 = (outf != nullptr);
#pragma unroll
    for (int mt = 0; mt < 4; ++mt) {
#pragma unroll
        for (int nt = 0; nt < 4; ++nt) {
            const int ncol = n0 + nt * 16 + lm;
#pragma unroll
            for (int r = 0; r < 8; ++r) {
                float v = acc[mt][nt][r];
                v = (v > 0.0f) ? v : v * NEG_SLOPE;
                const int m = m0 + mt * 16 + mrow + r;
                const size_t off = (size_t)m * NDIM + ncol;
                SoutG[off] = (__bf16)v;
                if (wantF32) outG[off] = v;
            }
        }
    }
}

extern "C" void kernel_launch(void* const* d_in, const int* in_sizes, int n_in,
                              void* d_out, int out_size, void* d_ws, size_t ws_size,
                              hipStream_t stream) {
    const float* x = (const float*)d_in[0];   // [2048, 512]
    const float* W = (const float*)d_in[1];   // [4096, 4096]
    float* out = (float*)d_out;               // [2048, 4096]

    // Workspace: Sa (16MB bf16) | Sb (16MB bf16) | Wt (32MB bf16) = 64MB.
    char* ws = (char*)d_ws;
    const size_t sbytes = (size_t)NBATCH * NDIM * sizeof(__bf16);
    __bf16* Sa = (__bf16*)(ws);
    __bf16* Sb = (__bf16*)(ws + sbytes);
    __bf16* Wt = (__bf16*)(ws + 2 * sbytes);

    // One-time (per launch) conversions.
    prep_wt<<<dim3(NDIM / 32, NDIM / 32), dim3(32, 8), 0, stream>>>(W, Wt);
    prep_s0<<<(NBATCH * NDIM) / 256, 256, 0, stream>>>(x, Sa);

    // 16 serialized GEMM+activation steps (stream ordering = dependency).
    for (int t = 0; t < TSTEPS; ++t) {
        const __bf16* in = (t & 1) ? Sb : Sa;
        __bf16* o        = (t & 1) ? Sa : Sb;
        float* fo        = (t == TSTEPS - 1) ? out : nullptr;
        reservoir_step<<<dim3(NDIM / 256, NBATCH / 128), 256, 0, stream>>>(
            in, Wt, o, fo);
    }
}
